// SwinBlock_41120016892062
// MI455X (gfx1250) — compile-verified, matrix operand order checked by reading, not run on hardware
//
#include <hip/hip_runtime.h>
#include <hip/hip_bf16.h>

// ---------------------------------------------------------------------------
// MI455X (gfx1250, wave32) fused Swin-quirk block.
// All matmuls on v_wmma_f32_16x16x32_bf16 (fp32 accumulate).
// Data movement: async global->LDS (ASYNCcnt) with explicit double buffering
// in the GEMM K-loops; Tensor Data Mover (TENSORcnt) for the contiguous
// 8KB Q/K/V tile fetches in the attention kernel. Both guarded so the code
// still compiles (with plain VGPR staging) on toolchains lacking the builtins.
//
// Fragment convention (wave32, 16x16x32 bf16):
//   lane L: r = L&15, half = L>>4
//   A (16xK row-major, row r) / B (KxN col-major, col r):
//     element e -> K = (e&7) + half*8 + (e>>3)*16
//   i.e. two contiguous 16B chunks at k-offsets half*8 and half*8+16.
//   C/D f32: element v -> M = v + 8*half, N = r.
// ---------------------------------------------------------------------------

typedef __bf16 bf16;
typedef __attribute__((ext_vector_type(16))) __bf16 bf16x16;
typedef __attribute__((ext_vector_type(8)))  __bf16 bf16x8;
typedef __attribute__((ext_vector_type(8)))  float  f32x8;
typedef __attribute__((ext_vector_type(4)))  unsigned int u32x4;
typedef __attribute__((ext_vector_type(4)))  int i32x4;
typedef __attribute__((ext_vector_type(8)))  int i32x8;

typedef __attribute__((address_space(3))) void lds_void_t;
typedef __attribute__((address_space(1))) i32x4 glb_i32x4;   // global 16B chunk
typedef __attribute__((address_space(3))) i32x4 lds_i32x4;   // LDS    16B chunk

#if defined(__has_builtin)
#if __has_builtin(__builtin_amdgcn_global_load_async_to_lds_b128) && \
    __has_builtin(__builtin_amdgcn_s_wait_asynccnt)
#define HAVE_ASYNC_LDS 1
#endif
#if __has_builtin(__builtin_amdgcn_tensor_load_to_lds) && \
    __has_builtin(__builtin_amdgcn_s_wait_tensorcnt)
#define HAVE_TDM 1
#endif
#endif

union FragU { bf16x16 v; bf16x8 h[2]; };

__device__ __forceinline__ bf16x16 ld_frag(const bf16* p) {
  // p points at (row base + half*8) elements; second chunk is +16 elements.
  FragU f;
  f.h[0] = *(const bf16x8*)(p);
  f.h[1] = *(const bf16x8*)(p + 16);
  return f.v;
}

__device__ __forceinline__ f32x8 wmma_bf16(bf16x16 a, bf16x16 b, f32x8 c) {
  return __builtin_amdgcn_wmma_f32_16x16x32_bf16(false, a, false, b, (short)0, c,
                                                 false, false);
}

// 16-byte global->LDS copy: async (direct-to-LDS, ASYNCcnt) when available.
__device__ __forceinline__ void cp16(bf16* lds_dst, const bf16* gsrc) {
#if defined(HAVE_ASYNC_LDS)
  __builtin_amdgcn_global_load_async_to_lds_b128((glb_i32x4*)gsrc,
                                                 (lds_i32x4*)lds_dst, 0, 0);
#else
  *(bf16x8*)lds_dst = *(const bf16x8*)gsrc;
#endif
}

__device__ __forceinline__ void cp_join() {
#if defined(HAVE_ASYNC_LDS)
  __builtin_amdgcn_s_wait_asynccnt(0);
#endif
}

#if defined(HAVE_TDM)
// 1-D TDM descriptor: copy 8192 contiguous bytes global->LDS.
// data_size=8B, tensor_dim0=tile_dim0=1024 elements, groups 2/3 unused.
__device__ __forceinline__ void tdm_load_8k(void* lds_dst, const void* gsrc) {
  unsigned lds = (unsigned)(unsigned long long)(lds_void_t*)lds_dst;
  unsigned long long ga = (unsigned long long)gsrc;
  u32x4 g0 = { 1u,                                    // count=1, user descriptor
               lds,                                   // lds_addr
               (unsigned)ga,                          // global_addr[31:0]
               (unsigned)((ga >> 32) & 0x01FFFFFFu) | 0x80000000u };  // type=2
  i32x8 g1 = { 0x00030000,            // data_size=3 (8B), no multicast/pad
               0x04000000,            // tensor_dim0[15:0]=1024 << 16
               0x00010000,            // tensor_dim0[31:16]=0, tensor_dim1=1
               (int)(1024u << 16),    // tile_dim0=1024
               0,                     // tile_dim1=0, tile_dim2=0
               1024,                  // tensor_dim0_stride=1024
               0, 0 };
  i32x4 z4 = { 0, 0, 0, 0 };
#if __clang_major__ >= 23
  i32x8 z8 = { 0, 0, 0, 0, 0, 0, 0, 0 };
  __builtin_amdgcn_tensor_load_to_lds(g0, g1, z4, z4, z8, 0);
#else
  __builtin_amdgcn_tensor_load_to_lds(g0, g1, z4, z4, 0);
#endif
}
#endif

// ---------------------------------------------------------------------------
// Prep kernels
// ---------------------------------------------------------------------------
__global__ void cvt_bf16_kernel(const float* __restrict__ src, bf16* __restrict__ dst, int n) {
  int i = blockIdx.x * 256 + threadIdx.x;
  if (i < n) dst[i] = (bf16)src[i];
}

__global__ void bn_prep_kernel(const float* __restrict__ g, const float* __restrict__ b,
                               const float* __restrict__ m, const float* __restrict__ v,
                               float* __restrict__ s, float* __restrict__ t) {
  int i = threadIdx.x;  // 256 channels
  float sc = g[i] * rsqrtf(v[i] + 1e-5f);
  s[i] = sc;
  t[i] = b[i] - m[i] * sc;
}

// ---------------------------------------------------------------------------
// BN1 + window partition into XmatT[b][pp=(s*256+c)][n]  (bf16, K=n contiguous)
// grid (64 s, 256 c, 8 b), block 256 (= n)
// ---------------------------------------------------------------------------
__global__ void bn1_pack_kernel(const float* __restrict__ x, const float* __restrict__ s1,
                                const float* __restrict__ t1, bf16* __restrict__ XT) {
  int n = threadIdx.x;          // window index
  int s = blockIdx.x;           // in-window position
  int c = blockIdx.y;           // channel
  int b = blockIdx.z;
  int h0 = n >> 4, w0 = n & 15;
  int y  = h0 * 8 + (s >> 3);
  int xw = w0 * 8 + (s & 7);
  float val = x[(((size_t)(b * 256 + c) * 128) + y) * 128 + xw];
  val = val * s1[c] + t1[c];
  XT[((size_t)b * 16384 + (s << 8) + c) * 256 + n] = (bf16)val;
}

// ---------------------------------------------------------------------------
// qkv GEMM: per b, (768 x 256) x (256 x 16384); double-buffered async K-loop;
// epilogue scatters into head-major Q[b,h,n2][s2][d], K[...][t2][d],
// Vt[...][d][t2] (bf16).  block 256 (8 waves); tile 64M x 128N; grid (128,12,8)
// ---------------------------------------------------------------------------
__global__ void qkv_gemm_kernel(const bf16* __restrict__ Wq, const float* __restrict__ qb,
                                const bf16* __restrict__ XT,
                                bf16* __restrict__ Q, bf16* __restrict__ Kk,
                                bf16* __restrict__ Vt) {
  __shared__ bf16 sA[2][64][32];
  __shared__ bf16 sB[2][128][32];
  const int b  = blockIdx.z;
  const int m0 = blockIdx.y * 64;
  const int n0 = blockIdx.x * 128;
  const int tid = threadIdx.x;
  const int lane = tid & 31, wave = tid >> 5;
  const int half = lane >> 4, lr = lane & 15;
  const int wm = (wave & 3) * 16;
  const int wn = (wave >> 2) * 64;
  const bf16* Xb = XT + (size_t)b * 16384 * 256;

  const int arow = tid >> 2, akk = (tid & 3) * 8;
  auto stage = [&](int buf, int k0) {
    cp16(&sA[buf][arow][akk], Wq + (size_t)(m0 + arow) * 256 + k0 + akk);
    for (int i = 0; i < 2; ++i) {
      int cch = tid + i * 256;
      int col = cch >> 2, kk = (cch & 3) * 8;
      cp16(&sB[buf][col][kk], Xb + (size_t)(n0 + col) * 256 + k0 + kk);
    }
  };

  f32x8 acc[4] = {};
  stage(0, 0);
  for (int it = 0; it < 8; ++it) {
    const int cur = it & 1;
    cp_join();
    __syncthreads();
    if (it < 7) stage(cur ^ 1, (it + 1) * 32);
    bf16x16 a = ld_frag(&sA[cur][wm + lr][half * 8]);
    for (int t = 0; t < 4; ++t) {
      bf16x16 bb = ld_frag(&sB[cur][wn + t * 16 + lr][half * 8]);
      acc[t] = wmma_bf16(a, bb, acc[t]);
    }
  }
  for (int t = 0; t < 4; ++t) {
    for (int v = 0; v < 8; ++v) {
      int o  = m0 + wm + v + half * 8;          // 0..767
      int pp = n0 + wn + t * 16 + lr;           // 0..16383
      float val = acc[t][v] + qb[o];
      int part = o >> 8, oc = o & 255;
      int head = oc >> 6, d = oc & 63;
      int s = pp >> 8, c = pp & 255;
      int ws_h = s >> 3, hh = s & 7;
      int ws_w = c >> 5, ww = c & 31;
      int n2 = hh * 32 + ww;
      int s2 = ws_h * 8 + ws_w;
      size_t gbase = ((size_t)((b * 4 + head) * 256 + n2)) << 12;  // *4096
      bf16 bv = (bf16)val;
      if (part == 0)      Q [gbase + s2 * 64 + d] = bv;
      else if (part == 1) Kk[gbase + s2 * 64 + d] = bv;
      else                Vt[gbase + d * 64 + s2] = bv;
    }
  }
}

// ---------------------------------------------------------------------------
// Attention: one workgroup (128 thr = 4 waves) per (b, h, n2): S = QK^T * 1/8,
// softmax rows, O = P V; scatter O into OT[b][pconv][ch] (proj B operand).
// Q/K/Vt tile fetch via TDM (one wave issues 3 descriptors) when available.
// grid (256 n2, 4 h, 8 b)
// ---------------------------------------------------------------------------
__global__ void attn_kernel(const bf16* __restrict__ Q, const bf16* __restrict__ Kk,
                            const bf16* __restrict__ Vt, bf16* __restrict__ OT) {
  __shared__ bf16 sQ[64 * 64];
  __shared__ bf16 sK[64 * 64];
  __shared__ bf16 sV[64 * 64];  // Vt: [d][t2]
  __shared__ bf16 sP[64 * 64];  // [s2][t2]
  const int n2 = blockIdx.x, h = blockIdx.y, b = blockIdx.z;
  const size_t base = ((size_t)((b * 4 + h) * 256 + n2)) << 12;
  const int tid = threadIdx.x;
  const int lane = tid & 31, wave = tid >> 5;
  const int half = lane >> 4, lr = lane & 15;

#if defined(HAVE_TDM)
  if (wave == 0) {
    tdm_load_8k(sQ, Q  + base);
    tdm_load_8k(sK, Kk + base);
    tdm_load_8k(sV, Vt + base);
    __builtin_amdgcn_s_wait_tensorcnt(0);
  }
#else
  for (int i = 0; i < 4; ++i) {  // 512 x 16B chunks each
    int c = tid + i * 128;
    cp16(&sQ[c * 8], Q  + base + c * 8);
    cp16(&sK[c * 8], Kk + base + c * 8);
    cp16(&sV[c * 8], Vt + base + c * 8);
  }
  cp_join();
#endif
  __syncthreads();

  const int wm = wave * 16;
  f32x8 sacc[4] = {};
  for (int k0 = 0; k0 < 64; k0 += 32) {
    bf16x16 a = ld_frag(&sQ[(wm + lr) * 64 + k0 + half * 8]);
    for (int t = 0; t < 4; ++t) {
      bf16x16 bb = ld_frag(&sK[(t * 16 + lr) * 64 + k0 + half * 8]);
      sacc[t] = wmma_bf16(a, bb, sacc[t]);
    }
  }
  const float scale = 0.125f;  // 64^-0.5
  for (int v = 0; v < 8; ++v) {
    float xv[4];
    for (int t = 0; t < 4; ++t) xv[t] = sacc[t][v] * scale;
    float mx = fmaxf(fmaxf(xv[0], xv[1]), fmaxf(xv[2], xv[3]));
    for (int off = 8; off >= 1; off >>= 1) mx = fmaxf(mx, __shfl_xor(mx, off, 32));
    float e[4], sum = 0.f;
    for (int t = 0; t < 4; ++t) { e[t] = __expf(xv[t] - mx); sum += e[t]; }
    for (int off = 8; off >= 1; off >>= 1) sum += __shfl_xor(sum, off, 32);
    float inv = 1.0f / sum;
    int row = wm + v + half * 8;
    for (int t = 0; t < 4; ++t) sP[row * 64 + t * 16 + lr] = (bf16)(e[t] * inv);
  }
  __syncthreads();
  f32x8 oacc[4] = {};
  for (int k0 = 0; k0 < 64; k0 += 32) {
    bf16x16 a = ld_frag(&sP[(wm + lr) * 64 + k0 + half * 8]);
    for (int t = 0; t < 4; ++t) {
      bf16x16 bb = ld_frag(&sV[(t * 16 + lr) * 64 + k0 + half * 8]);
      oacc[t] = wmma_bf16(a, bb, oacc[t]);
    }
  }
  const int hh = n2 >> 5, ww = n2 & 31;
  for (int t = 0; t < 4; ++t) {
    for (int v = 0; v < 8; ++v) {
      int s2 = wm + v + half * 8;
      int d  = t * 16 + lr;
      int ws_h = s2 >> 3, ws_w = s2 & 7;
      int sc = hh * 8 + ws_h;
      int cc = ww * 8 + ws_w;
      int ch = h * 64 + d;
      OT[((size_t)b * 16384 + sc * 256 + cc) * 256 + ch] = (bf16)oacc[t][v];
    }
  }
}

// ---------------------------------------------------------------------------
// proj GEMM (256x256)x(256x16384) per b; double-buffered async K-loop;
// epilogue: window_reverse + shortcut residual -> hbuf (fp32), and
// BN2 -> bnhT[b][pix][c] (bf16, MLP B operand).
// block 256; tile 64M x 128N; grid (128, 4, 8)
// ---------------------------------------------------------------------------
__global__ void proj_gemm_kernel(const bf16* __restrict__ Wp, const float* __restrict__ pb,
                                 const bf16* __restrict__ OT, const float* __restrict__ xin,
                                 const float* __restrict__ s2v, const float* __restrict__ t2v,
                                 float* __restrict__ hbuf, bf16* __restrict__ bnhT) {
  __shared__ bf16 sA[2][64][32];
  __shared__ bf16 sB[2][128][32];
  const int b  = blockIdx.z;
  const int m0 = blockIdx.y * 64;
  const int n0 = blockIdx.x * 128;
  const int tid = threadIdx.x;
  const int lane = tid & 31, wave = tid >> 5;
  const int half = lane >> 4, lr = lane & 15;
  const int wm = (wave & 3) * 16;
  const int wn = (wave >> 2) * 64;
  const bf16* Ob = OT + (size_t)b * 16384 * 256;

  const int arow = tid >> 2, akk = (tid & 3) * 8;
  auto stage = [&](int buf, int k0) {
    cp16(&sA[buf][arow][akk], Wp + (size_t)(m0 + arow) * 256 + k0 + akk);
    for (int i = 0; i < 2; ++i) {
      int cch = tid + i * 256;
      int col = cch >> 2, kk = (cch & 3) * 8;
      cp16(&sB[buf][col][kk], Ob + (size_t)(n0 + col) * 256 + k0 + kk);
    }
  };

  f32x8 acc[4] = {};
  stage(0, 0);
  for (int it = 0; it < 8; ++it) {
    const int cur = it & 1;
    cp_join();
    __syncthreads();
    if (it < 7) stage(cur ^ 1, (it + 1) * 32);
    bf16x16 a = ld_frag(&sA[cur][wm + lr][half * 8]);
    for (int t = 0; t < 4; ++t) {
      bf16x16 bb = ld_frag(&sB[cur][wn + t * 16 + lr][half * 8]);
      acc[t] = wmma_bf16(a, bb, acc[t]);
    }
  }
  for (int t = 0; t < 4; ++t) {
    for (int v = 0; v < 8; ++v) {
      int o  = m0 + wm + v + half * 8;   // proj out channel == window index
      int pp = n0 + wn + t * 16 + lr;
      float val = acc[t][v] + pb[o];
      int s = pp >> 8, c = pp & 255;
      int y  = (o >> 4) * 8 + (s >> 3);
      int xw = (o & 15) * 8 + (s & 7);
      size_t idx = (((size_t)(b * 256 + c) * 128) + y) * 128 + xw;
      float hv = xin[idx] + val;
      hbuf[idx] = hv;
      float bn = hv * s2v[c] + t2v[c];
      bnhT[((size_t)b * 16384 + y * 128 + xw) * 256 + c] = (bf16)bn;
    }
  }
}

// ---------------------------------------------------------------------------
// Fused MLP: per workgroup 64 pixels. LDS: sX 64x256 bf16 (32KB) +
// sM (m1^T, GELU'd) 64x1024 bf16 (128KB) = 160KB dynamic LDS (CDNA5 allows
// up to 320KB/workgroup). W1/W2 fragments stream from global (L2-resident),
// with global_prefetch hints one K-step ahead.
// block 256 (8 waves); grid (256, 8)
// ---------------------------------------------------------------------------
__global__ void mlp_kernel(const bf16* __restrict__ W1, const float* __restrict__ b1,
                           const bf16* __restrict__ W2, const float* __restrict__ b2,
                           const bf16* __restrict__ XT, const float* __restrict__ hbuf,
                           float* __restrict__ out) {
  extern __shared__ bf16 smem[];
  bf16* sX = smem;              // [64][256]
  bf16* sM = smem + 64 * 256;   // [64][1024]
  const int b  = blockIdx.y;
  const int p0 = blockIdx.x * 64;
  const int tid = threadIdx.x;
  const bf16* Xb = XT + ((size_t)b * 16384 + p0) * 256;
  for (int i = 0; i < 8; ++i) {  // 2048 x 16B chunks
    int c = tid + i * 256;
    cp16(&sX[c * 8], Xb + c * 8);
  }
  cp_join();
  __syncthreads();
  const int lane = tid & 31, wave = tid >> 5;
  const int half = lane >> 4, lr = lane & 15;

  // GEMM1: m1 = gelu(W1 (1024x256) @ X) -> sM as [p][r]
  for (int mb = 0; mb < 8; ++mb) {
    int mrow = mb * 128 + wave * 16;
    const bf16* w1row = W1 + (size_t)(mrow + lr) * 256;
    f32x8 acc[4] = {};
    for (int k0 = 0; k0 < 256; k0 += 32) {
      if (k0 + 32 < 256) __builtin_prefetch(w1row + k0 + 32, 0, 0);
      bf16x16 a = ld_frag(w1row + k0 + half * 8);
      for (int t = 0; t < 4; ++t) {
        bf16x16 bb = ld_frag(&sX[(t * 16 + lr) * 256 + k0 + half * 8]);
        acc[t] = wmma_bf16(a, bb, acc[t]);
      }
    }
    for (int t = 0; t < 4; ++t) {
      for (int v = 0; v < 8; ++v) {
        int r = mrow + v + half * 8;
        int p = t * 16 + lr;
        float xv = acc[t][v] + b1[r];
        float g = 0.5f * xv * (1.0f + erff(xv * 0.70710678118f));  // exact GELU
        sM[p * 1024 + r] = (bf16)g;
      }
    }
  }
  __syncthreads();

  // GEMM2: m2 = W2 (256x1024) @ m1 ; out = hbuf + m2 + b2
  const bf16* w2r0 = W2 + (size_t)(wave * 32 + lr) * 1024;
  const bf16* w2r1 = W2 + (size_t)(wave * 32 + 16 + lr) * 1024;
  f32x8 acc2[2][4] = {};
  for (int k0 = 0; k0 < 1024; k0 += 32) {
    if (k0 + 32 < 1024) {
      __builtin_prefetch(w2r0 + k0 + 32, 0, 0);
      __builtin_prefetch(w2r1 + k0 + 32, 0, 0);
    }
    bf16x16 a0 = ld_frag(w2r0 + k0 + half * 8);
    bf16x16 a1 = ld_frag(w2r1 + k0 + half * 8);
    for (int t = 0; t < 4; ++t) {
      bf16x16 bb = ld_frag(&sM[(t * 16 + lr) * 1024 + k0 + half * 8]);
      acc2[0][t] = wmma_bf16(a0, bb, acc2[0][t]);
      acc2[1][t] = wmma_bf16(a1, bb, acc2[1][t]);
    }
  }
  for (int mi = 0; mi < 2; ++mi) {
    for (int t = 0; t < 4; ++t) {
      for (int v = 0; v < 8; ++v) {
        int cch = wave * 32 + mi * 16 + v + half * 8;
        int p   = p0 + t * 16 + lr;
        size_t idx = ((size_t)(b * 256 + cch)) * 16384 + p;
        out[idx] = hbuf[idx] + acc2[mi][t][v] + b2[cch];
      }
    }
  }
}

// ---------------------------------------------------------------------------
extern "C" void kernel_launch(void* const* d_in, const int* in_sizes, int n_in,
                              void* d_out, int out_size, void* d_ws, size_t ws_size,
                              hipStream_t stream) {
  const float* x      = (const float*)d_in[0];
  const float* qkv_w  = (const float*)d_in[1];
  const float* qkv_b  = (const float*)d_in[2];
  const float* proj_w = (const float*)d_in[3];
  const float* proj_b = (const float*)d_in[4];
  const float* mlp_w1 = (const float*)d_in[5];
  const float* mlp_b1 = (const float*)d_in[6];
  const float* mlp_w2 = (const float*)d_in[7];
  const float* mlp_b2 = (const float*)d_in[8];
  const float* bn1_g = (const float*)d_in[9];
  const float* bn1_b = (const float*)d_in[10];
  const float* bn1_m = (const float*)d_in[11];
  const float* bn1_v = (const float*)d_in[12];
  const float* bn2_g = (const float*)d_in[13];
  const float* bn2_b = (const float*)d_in[14];
  const float* bn2_m = (const float*)d_in[15];
  const float* bn2_v = (const float*)d_in[16];
  float* out = (float*)d_out;

  char* ws = (char*)d_ws;
  // Workspace layout (region reuse keeps peak ~386 MiB):
  bf16* Wqkv = (bf16*)(ws + 0);                    // 768*256
  bf16* Wprj = (bf16*)(ws + 393216);               // 256*256
  bf16* Wm1  = (bf16*)(ws + 524288);               // 1024*256
  bf16* Wm2  = (bf16*)(ws + 1048576);              // 256*1024
  float* s1  = (float*)(ws + 1572864);
  float* t1  = (float*)(ws + 1572864 + 1024);
  float* s2  = (float*)(ws + 1572864 + 2048);
  float* t2  = (float*)(ws + 1572864 + 3072);
  bf16* XT   = (bf16*)(ws + 2097152);              // 64 MiB  (later: OT)
  bf16* Qb   = (bf16*)(ws + 69206016);             // 64 MiB  (later: bnhT)
  bf16* Kb   = (bf16*)(ws + 136314880);            // 64 MiB
  bf16* Vtb  = (bf16*)(ws + 203423744);            // 64 MiB
  float* hbuf = (float*)(ws + 270532608);          // 128 MiB
  bf16* OT   = XT;    // XmatT consumed by qkv GEMM before attention writes OT
  bf16* bnhT = Qb;    // Q consumed by attention before proj writes bnhT

  // Prep: weights to bf16, fold BN into scale/shift
  cvt_bf16_kernel<<<(196608 + 255) / 256, 256, 0, stream>>>(qkv_w,  Wqkv, 196608);
  cvt_bf16_kernel<<<(65536  + 255) / 256, 256, 0, stream>>>(proj_w, Wprj, 65536);
  cvt_bf16_kernel<<<(262144 + 255) / 256, 256, 0, stream>>>(mlp_w1, Wm1,  262144);
  cvt_bf16_kernel<<<(262144 + 255) / 256, 256, 0, stream>>>(mlp_w2, Wm2,  262144);
  bn_prep_kernel<<<1, 256, 0, stream>>>(bn1_g, bn1_b, bn1_m, bn1_v, s1, t1);
  bn_prep_kernel<<<1, 256, 0, stream>>>(bn2_g, bn2_b, bn2_m, bn2_v, s2, t2);

  // Stage 1: BN1 + window-partition pack
  bn1_pack_kernel<<<dim3(64, 256, 8), 256, 0, stream>>>(x, s1, t1, XT);

  // Stage 2: qkv GEMM -> head-layout Q/K/Vt
  qkv_gemm_kernel<<<dim3(128, 12, 8), 256, 0, stream>>>(Wqkv, qkv_b, XT, Qb, Kb, Vtb);

  // Stage 3: 8192 independent 64x64x64 attentions
  attn_kernel<<<dim3(256, 4, 8), 128, 0, stream>>>(Qb, Kb, Vtb, OT);

  // Stage 4: proj GEMM + window_reverse + residual + BN2
  proj_gemm_kernel<<<dim3(128, 4, 8), 256, 0, stream>>>(Wprj, proj_b, OT, x, s2, t2,
                                                        hbuf, bnhT);

  // Stage 5: fused MLP (160 KB dynamic LDS per workgroup)
  size_t mlp_lds = (size_t)(64 * 256 + 64 * 1024) * sizeof(bf16);  // 163840 B
  mlp_kernel<<<dim3(256, 8), 256, mlp_lds, stream>>>(Wm1, mlp_b1, Wm2, mlp_b2,
                                                     bnhT, hbuf, out);
}
